// PostProcessor_22969485099775
// MI455X (gfx1250) — compile-verified, matrix operand order checked by reading, not run on hardware
//
#include <hip/hip_runtime.h>
#include <hip/hip_bf16.h>

typedef __attribute__((ext_vector_type(16))) _Float16 v16h;
typedef __attribute__((ext_vector_type(8)))  float    v8f;
typedef unsigned long long u64;
typedef unsigned int       u32;

#define NPROP 30000
#define NCLS  81           // classes incl background
#define KPOOL 256          // per-class NMS pool
#define NDET  100
#define CAP   8192         // per-class candidate cap (expected ~900)
#define NEGV  (-1e30f)
#define VALID_CUT (-5e29f) // NEG/2
#define IMGW_M1 1332.0f
#define IMGH_M1 799.0f
#define BBOX_CLIP 4.135166556742356f   // log(1000/16)
#define AREA_SCALE (1.0f/256.0f)       // keep f16 areas in range; IoU scale-invariant

// ---------------- decode helpers ----------------
__device__ __forceinline__ void refGeom(const float* __restrict__ props, int prop,
                                        float& w, float& h, float& cx, float& cy) {
    const float* pb = props + (size_t)prop * 4;
    float x1 = pb[0], y1 = pb[1], x2 = pb[2], y2 = pb[3];
    w = x2 - x1 + 1.0f;  h = y2 - y1 + 1.0f;
    cx = x1 + 0.5f * w;  cy = y1 + 0.5f * h;
}

__device__ __forceinline__ void decodeRect(const float* __restrict__ boxReg,
                                           const float* __restrict__ props,
                                           int prop, int label,
                                           float& bx1, float& by1, float& bx2, float& by2) {
    float w, h, cx, cy; refGeom(props, prop, w, h, cx, cy);
    const float* rr = boxReg + (size_t)prop * (4 * NCLS) + label * 4;
    float dx = rr[0] * 0.1f, dy = rr[1] * 0.1f;
    float dw = fminf(rr[2] * 0.2f, BBOX_CLIP);
    float dh = fminf(rr[3] * 0.2f, BBOX_CLIP);
    float pcx = dx * w + cx, pcy = dy * h + cy;
    float pw = __expf(dw) * w, ph = __expf(dh) * h;
    bx1 = pcx - 0.5f * pw;        by1 = pcy - 0.5f * ph;
    bx2 = pcx + 0.5f * pw - 1.0f; by2 = pcy + 0.5f * ph - 1.0f;
    bx1 = fminf(fmaxf(bx1, 0.0f), IMGW_M1);
    bx2 = fminf(fmaxf(bx2, 0.0f), IMGW_M1);
    by1 = fminf(fmaxf(by1, 0.0f), IMGH_M1);
    by2 = fminf(fmaxf(by2, 0.0f), IMGH_M1);
}

__device__ __forceinline__ void decodeQuad(const float* __restrict__ quadReg,
                                           const float* __restrict__ props,
                                           int prop, int label, float* q /*8*/) {
    float w, h, cx, cy; refGeom(props, prop, w, h, cx, cy);
    const float* rr = quadReg + (size_t)prop * (8 * NCLS) + label * 8;
    #pragma unroll
    for (int k = 0; k < 4; ++k) {
        float px = (rr[2 * k]     * 0.1f) * w + cx;
        float py = (rr[2 * k + 1] * 0.1f) * h + cy;
        q[2 * k]     = fminf(fmaxf(px, 0.0f), IMGW_M1);
        q[2 * k + 1] = fminf(fmaxf(py, 0.0f), IMGH_M1);
    }
}

// ---------------- kernel 0: zero per-class counters ----------------
__global__ void pp_zero_counters(int* __restrict__ cnt) {
    if (threadIdx.x < (NCLS - 1)) cnt[threadIdx.x] = 0;
}

// ---------------- kernel 1: softmax, transposed score store ----------------
__global__ __launch_bounds__(256) void pp_softmax(const float* __restrict__ logits,
                                                  float* __restrict__ scores /*[80][NPROP]*/) {
    int lane = threadIdx.x & 31;
    int row  = blockIdx.x * 8 + (threadIdx.x >> 5);
    if (row >= NPROP) return;
    const float* rp = logits + (size_t)row * NCLS;
    __builtin_prefetch(rp + 8 * NCLS, 0, 0);   // global_prefetch_b8
    float l0 = rp[lane];
    float l1 = rp[lane + 32];
    float l2 = (lane + 64 < NCLS) ? rp[lane + 64] : -3.4e38f;
    float m = fmaxf(l0, fmaxf(l1, l2));
    #pragma unroll
    for (int o = 16; o > 0; o >>= 1) m = fmaxf(m, __shfl_xor(m, o, 32));
    float e0 = __expf(l0 - m);
    float e1 = __expf(l1 - m);
    float e2 = (lane + 64 < NCLS) ? __expf(l2 - m) : 0.0f;
    float s = e0 + e1 + e2;
    #pragma unroll
    for (int o = 16; o > 0; o >>= 1) s += __shfl_xor(s, o, 32);
    float inv = 1.0f / s;
    if (lane >= 1) scores[(size_t)(lane - 1) * NPROP + row] = e0 * inv;   // classes 1..31
    scores[(size_t)(lane + 31) * NPROP + row] = e1 * inv;                 // classes 32..63
    if (lane + 64 < NCLS) scores[(size_t)(lane + 63) * NPROP + row] = e2 * inv; // 64..80
}

// ---------------- kernel 2: per-class threshold compaction ----------------
__global__ __launch_bounds__(256) void pp_compact(const float* __restrict__ scores,
                                                  u64* __restrict__ cand,
                                                  int* __restrict__ cnt) {
    int c = blockIdx.y;
    int n = blockIdx.x * blockDim.x + threadIdx.x;
    if (n >= NPROP) return;
    float s = scores[(size_t)c * NPROP + n];
    if (s > 0.05f) {
        int pos = atomicAdd(&cnt[c], 1);
        if (pos < CAP) {
            // unique key: (positive-float bits)<<32 | ~idx  => deterministic after sort,
            // ties broken toward smaller proposal index (like top_k).
            u64 key = ((u64)__float_as_uint(s) << 32) | (u32)(~(u32)n);
            cand[(size_t)c * CAP + pos] = key;
        }
    }
}

// ---------------- kernel 3: per-class bitonic sort (desc), emit top-256 ----------------
__global__ __launch_bounds__(1024) void pp_sort_topk(const u64* __restrict__ cand,
                                                     const int* __restrict__ cnt,
                                                     float* __restrict__ topScore,
                                                     int* __restrict__ topIdx) {
    __shared__ u64 keys[CAP];   // 64 KB LDS
    int c = blockIdx.x, t = threadIdx.x;
    int n = min(cnt[c], CAP);
    for (int i = t; i < CAP; i += 1024)
        keys[i] = (i < n) ? cand[(size_t)c * CAP + i] : 0ull;
    __syncthreads();
    for (int k = 2; k <= CAP; k <<= 1) {
        for (int j = k >> 1; j > 0; j >>= 1) {
            for (int i = t; i < CAP; i += 1024) {
                int ixj = i ^ j;
                if (ixj > i) {
                    u64 a = keys[i], b = keys[ixj];
                    bool up = ((i & k) == 0);
                    if ((a < b) == up) { keys[i] = b; keys[ixj] = a; }  // descending
                }
            }
            __syncthreads();
        }
    }
    if (t < KPOOL) {
        u64 key = keys[t];
        u32 hi = (u32)(key >> 32);
        topScore[c * KPOOL + t] = hi ? __uint_as_float(hi) : NEGV;
        topIdx[c * KPOOL + t]   = hi ? (int)(~(u32)key) : 0;
    }
}

// ---------------- kernel 4: per-class NMS with WMMA area-sum tiles ----------------
__global__ __launch_bounds__(256) void pp_nms(const float* __restrict__ topScore,
                                              const int* __restrict__ topIdx,
                                              const float* __restrict__ boxReg,
                                              const float* __restrict__ props,
                                              float* __restrict__ keptScore) {
    __shared__ float X1[KPOOL], Y1[KPOOL], X2[KPOOL], Y2[KPOOL];
    __shared__ _Float16 AH[KPOOL];
    __shared__ u32 sup[KPOOL * 8];          // 256 rows x 256 bits
    __shared__ unsigned char keep[KPOOL];

    int c = blockIdx.x, t = threadIdx.x;
    float s = topScore[c * KPOOL + t];
    bool valid = s > VALID_CUT;
    int prop = topIdx[c * KPOOL + t];
    float bx1 = 0.f, by1 = 0.f, bx2 = -1.f, by2 = -1.f;   // dummy => area 0
    if (valid) decodeRect(boxReg, props, prop, c + 1, bx1, by1, bx2, by2);
    X1[t] = bx1; Y1[t] = by1; X2[t] = bx2; Y2[t] = by2;
    float area = (bx2 - bx1 + 1.0f) * (by2 - by1 + 1.0f);
    AH[t] = (_Float16)(area * AREA_SCALE);
    #pragma unroll
    for (int w = 0; w < 8; ++w) sup[t * 8 + w] = 0u;
    keep[t] = valid ? 1 : 0;
    __syncthreads();

    int waveId = t >> 5, lane = t & 31;
    const _Float16 one = (_Float16)1.0f;
    // 256x256 IoU matrix in 16x16 tiles; area_i + area_j via rank-2 WMMA outer product.
    for (int tile = waveId; tile < 256; tile += 8) {
        int ti = tile >> 4, tj = tile & 15;
        v16h a = {}; v16h b = {};
        if (lane < 16) {
            a[0] = AH[ti * 16 + lane]; a[1] = one;       // A row: [area_i, 1, 0...]
            b[0] = one; b[1] = AH[tj * 16 + lane];       // B col: [1, area_j, 0...]
        }
        v8f acc = {};
        acc = __builtin_amdgcn_wmma_f32_16x16x32_f16(false, a, false, b,
                                                     (short)0, acc, false, false);
        int j = tj * 16 + (lane & 15);
        int base = (lane < 16) ? 0 : 8;
        float xj1 = X1[j], yj1 = Y1[j], xj2 = X2[j], yj2 = Y2[j];
        u32 bits = 0;
        #pragma unroll
        for (int r = 0; r < 8; ++r) {
            int i = ti * 16 + base + r;
            float iw = fminf(X2[i], xj2) - fmaxf(X1[i], xj1) + 1.0f;
            float ih = fminf(Y2[i], yj2) - fmaxf(Y1[i], yj1) + 1.0f;
            float inter = (iw > 0.0f && ih > 0.0f) ? iw * ih * AREA_SCALE : 0.0f;
            float iou = inter / (acc[r] - inter);        // scaled num & den cancel
            if (iou > 0.5f) bits |= 1u << (((ti & 1) * 16 + base + r) & 31);
        }
        if (bits) atomicOr(&sup[j * 8 + (ti >> 1)], bits);
    }
    __syncthreads();

    // greedy suppression: IoU matrix is symmetric, so column mask == row mask.
    for (int i = 0; i < 256; ++i) {
        __syncthreads();
        if (keep[i] && t > i && ((sup[t * 8 + (i >> 5)] >> (i & 31)) & 1u))
            keep[t] = 0;
    }
    __syncthreads();
    keptScore[c * KPOOL + t] = keep[t] ? s : NEGV;
}

// ---------------- kernel 5: global top-100 + lazy decode of winners ----------------
__global__ __launch_bounds__(1024) void pp_final(const float* __restrict__ keptScore,
                                                 const int* __restrict__ topIdx,
                                                 const float* __restrict__ boxReg,
                                                 const float* __restrict__ quadReg,
                                                 const float* __restrict__ props,
                                                 float* __restrict__ dets,
                                                 int* __restrict__ labels) {
    const int TOT = (NCLS - 1) * KPOOL;     // 20480
    __shared__ u64 red[1024];
    __shared__ u32 taken[TOT / 32];
    int t = threadIdx.x;
    for (int i = t; i < TOT / 32; i += 1024) taken[i] = 0u;
    __syncthreads();

    for (int r = 0; r < NDET; ++r) {
        u64 best = 0ull;
        for (int i = t; i < TOT; i += 1024) {
            if ((taken[i >> 5] >> (i & 31)) & 1u) continue;
            float s = keptScore[i];
            u32 b = __float_as_uint(s);
            u32 o = (b & 0x80000000u) ? ~b : (b | 0x80000000u);   // order-preserving
            u64 key = ((u64)o << 32) | (u32)(~(u32)i);            // ties: smaller i wins
            if (key > best) best = key;
        }
        red[t] = best;
        __syncthreads();
        for (int sft = 512; sft > 0; sft >>= 1) {
            if (t < sft) { u64 a = red[t], b = red[t + sft]; red[t] = (a > b) ? a : b; }
            __syncthreads();
        }
        u64 wk = red[0];
        int flat = (int)(~(u32)wk);
        if (t == 0) {
            taken[flat >> 5] |= 1u << (flat & 31);
            float s = keptScore[flat];
            bool ok = s > VALID_CUT;
            int cls = flat >> 8;
            int prop = topIdx[flat];
            int label = cls + 1;
            float o13[13];
            if (ok) {
                float bx1, by1, bx2, by2;
                decodeRect(boxReg, props, prop, label, bx1, by1, bx2, by2);
                o13[0] = bx1; o13[1] = by1; o13[2] = bx2; o13[3] = by2;
                decodeQuad(quadReg, props, prop, label, &o13[4]);
                o13[12] = s;
            } else {
                #pragma unroll
                for (int k = 0; k < 13; ++k) o13[k] = 0.0f;
                label = 0;
            }
            #pragma unroll
            for (int k = 0; k < 13; ++k) dets[r * 13 + k] = o13[k];
            labels[r] = label;
        }
        __syncthreads();
    }
}

extern "C" void kernel_launch(void* const* d_in, const int* in_sizes, int n_in,
                              void* d_out, int out_size, void* d_ws, size_t ws_size,
                              hipStream_t stream) {
    const float* logits  = (const float*)d_in[0];   // [30000, 81]
    const float* boxReg  = (const float*)d_in[1];   // [30000, 324]
    const float* quadReg = (const float*)d_in[2];   // [30000, 648]
    const float* props   = (const float*)d_in[3];   // [30000, 4]

    char* ws = (char*)d_ws;
    size_t off = 0;
    float* scores   = (float*)(ws + off); off += (size_t)(NCLS - 1) * NPROP * 4;  // 9.6 MB
    int*   cnt      = (int*)  (ws + off); off += 512;                             // 80 cnt, padded
    u64*   cand     = (u64*)  (ws + off); off += (size_t)(NCLS - 1) * CAP * 8;    // 5.24 MB
    float* topScore = (float*)(ws + off); off += (size_t)(NCLS - 1) * KPOOL * 4;
    int*   topIdx   = (int*)  (ws + off); off += (size_t)(NCLS - 1) * KPOOL * 4;
    float* keptScore= (float*)(ws + off); off += (size_t)(NCLS - 1) * KPOOL * 4;

    pp_zero_counters<<<1, 128, 0, stream>>>(cnt);
    pp_softmax<<<NPROP / 8, 256, 0, stream>>>(logits, scores);
    pp_compact<<<dim3((NPROP + 255) / 256, NCLS - 1), 256, 0, stream>>>(scores, cand, cnt);
    pp_sort_topk<<<NCLS - 1, 1024, 0, stream>>>(cand, cnt, topScore, topIdx);
    pp_nms<<<NCLS - 1, 256, 0, stream>>>(topScore, topIdx, boxReg, props, keptScore);
    pp_final<<<1, 1024, 0, stream>>>(keptScore, topIdx, boxReg, quadReg, props,
                                     (float*)d_out, (int*)d_out + NDET * 13);
}
